// NSMCell_31731218382972
// MI455X (gfx1250) — matrix operand order repaired and checked by reading, not compiled, not run on hardware
//
#include <hip/hip_runtime.h>

typedef __attribute__((ext_vector_type(16))) __bf16 v16bf;
typedef __attribute__((ext_vector_type(8)))  __bf16 v8bf;
typedef __attribute__((ext_vector_type(8)))  float  v8f;
typedef __attribute__((ext_vector_type(4)))  float  v4f;

constexpr int B = 32, N = 512, P = 8, H = 300;
constexpr int KT  = 32;                   // K per bf16 WMMA step
constexpr int NKT = (H + KT - 1) / KT;    // 10 k-steps (last partial, zero-padded)
constexpr int NHT = (H + 15) / 16;        // 19 h-tiles of 16 rows
constexpr int HP  = 304;                  // padded h rows in Wt (multiple of 16)
constexpr int KP  = 320;                  // padded k in Wt (covers 10*32, zero tail)

// ---------------------------------------------------------------------------
// Prep: Wt[p][h][k] = bf16(W[p][k][h]), transposed + zero-padded.
// ---------------------------------------------------------------------------
__global__ __launch_bounds__(256)
void nsm_prep_kernel(const float* __restrict__ Wn, __bf16* __restrict__ Wt)
{
    const int idx = blockIdx.x * 256 + threadIdx.x;   // P*HP*KP = 778240 total
    const int p = idx / (HP * KP);
    const int r = idx % (HP * KP);
    const int h = r / KP;
    const int k = r % KP;
    float v = 0.f;
    if (h < H && k < H) v = Wn[(p * H + k) * H + h];
    Wt[idx] = (__bf16)v;
}

// ---------------------------------------------------------------------------
// Process NJ h-tiles simultaneously for one lane's node n: one B-fragment
// build (node_attr load + instruction scale + bf16 pack) feeds NJ WMMAs.
// Returns this lane's partial score contribution sum_h elu(s)*w_state[h].
// ---------------------------------------------------------------------------
template <int NJ>
__device__ __forceinline__
float tile_pass(const int* h0w,
                const float* __restrict__ nrow0,   // node_attr row base (p-stride H)
                const float* __restrict__ irow,    // instruction[b,:]
                const float* __restrict__ npsrow,  // npsim[b,:]
                const __bf16* __restrict__ Wt,
                const float* __restrict__ w_state,
                int lm, int half)
{
    v8f sum[NJ], sumsq[NJ];
    #pragma unroll
    for (int t = 0; t < NJ; ++t) { sum[t] = {}; sumsq[t] = {}; }

    for (int p = 0; p < P; ++p) {
        const float nps = npsrow[p];
        const float* nrow = nrow0 + (size_t)p * H;
        const __bf16* arow[NJ];
        #pragma unroll
        for (int t = 0; t < NJ; ++t)
            arow[t] = Wt + ((size_t)(p * HP + h0w[t] + lm)) * KP;

        v8f acc[NJ];
        #pragma unroll
        for (int t = 0; t < NJ; ++t) acc[t] = {};

        #pragma unroll
        for (int kt = 0; kt < NKT; ++kt) {
            const int k0 = kt * KT;
            const int kb = k0 + half * 16;   // B element e -> K = kb + e (contiguous)

            float vals[16];
            if (kt < NKT - 1) {              // compile-time split: full 16-wide chunk
                const v4f x0 = *(const v4f*)(nrow + kb);
                const v4f x1 = *(const v4f*)(nrow + kb + 4);
                const v4f x2 = *(const v4f*)(nrow + kb + 8);
                const v4f x3 = *(const v4f*)(nrow + kb + 12);
                const v4f i0 = *(const v4f*)(irow + kb);
                const v4f i1 = *(const v4f*)(irow + kb + 4);
                const v4f i2 = *(const v4f*)(irow + kb + 8);
                const v4f i3 = *(const v4f*)(irow + kb + 12);
                #pragma unroll
                for (int e = 0; e < 4; ++e) {
                    vals[e]      = x0[e] * i0[e];
                    vals[e + 4]  = x1[e] * i1[e];
                    vals[e + 8]  = x2[e] * i2[e];
                    vals[e + 12] = x3[e] * i3[e];
                }
            } else {                         // K tail (k0 = 288): guarded loads
                #pragma unroll
                for (int e = 0; e < 16; ++e) {
                    const int k = kb + e;
                    vals[e] = (k < H) ? nrow[k] * irow[k] : 0.f;
                }
            }
            v16bf bf;
            #pragma unroll
            for (int e = 0; e < 16; ++e) bf[e] = (__bf16)vals[e];

            // NJ A-fragments (Wt): element e -> K = k0 + half*16 + e (contiguous)
            #pragma unroll
            for (int t = 0; t < NJ; ++t) {
                v8bf alo = *(const v8bf*)(arow[t] + k0 + half * 16);
                v8bf ahi = *(const v8bf*)(arow[t] + k0 + half * 16 + 8);
                v16bf a = __builtin_shufflevector(alo, ahi,
                    0, 1, 2, 3, 4, 5, 6, 7, 8, 9, 10, 11, 12, 13, 14, 15);
                acc[t] = __builtin_amdgcn_wmma_f32_16x16x32_bf16(
                    false, a, false, bf, (short)0, acc[t], false, false);
            }
        }
        #pragma unroll
        for (int t = 0; t < NJ; ++t)
            #pragma unroll
            for (int i = 0; i < 8; ++i) {
                const float v = acc[t][i] * nps;
                sum[t][i]   += v;
                sumsq[t][i] += v * v;
            }
    }

    float partial = 0.f;
    #pragma unroll
    for (int t = 0; t < NJ; ++t)
        #pragma unroll
        for (int i = 0; i < 8; ++i) {
            const int h = h0w[t] + half * 8 + i;          // D row M = half*8 + i
            const float wcoef = (h < H) ? w_state[h] : 0.f;
            const float nrm = sqrtf(sumsq[t][i]);
            const float s   = sum[t][i] / fmaxf(nrm, 1e-12f);
            const float e   = (s > 0.f) ? s : (__expf(s) - 1.f);
            partial += e * wcoef;
        }
    return partial;
}

// ---------------------------------------------------------------------------
// Main fused kernel: one block = 16-node slab of one batch; 8 waves cover all
// 19 h-tiles (2 tiles simultaneously, +1 leftover tile for waves 0..2).
// ---------------------------------------------------------------------------
__global__ __launch_bounds__(256)
void nsm_gemm_kernel(const float* __restrict__ node_attr,
                     const float* __restrict__ instruction,
                     const float* __restrict__ npsim,
                     const __bf16* __restrict__ Wt,
                     const float* __restrict__ w_state,
                     float* __restrict__ scores)
{
    __shared__ float sc[8][16];

    const int tid  = threadIdx.x;
    const int lane = tid & 31;
    const int wave = tid >> 5;
    const int lm   = lane & 15;
    const int half = lane >> 4;

    const int b  = blockIdx.y;
    const int n0 = blockIdx.x * 16;
    const int n  = n0 + lm;

    const float* irow   = instruction + b * H;
    const float* npsrow = npsim + b * P;
    const float* nrow0  = node_attr + ((size_t)(b * N + n)) * P * H;

    // pass 1: two guaranteed tiles per wave, one B-build feeds two WMMAs
    const int h2[2] = { wave * 16, (wave + 8) * 16 };
    float partial = tile_pass<2>(h2, nrow0, irow, npsrow, Wt, w_state, lm, half);

    // pass 2: leftover tiles 16..18 (waves 0..2)
    if (wave < NHT - 16) {
        const int h1[1] = { (16 + wave) * 16 };
        partial += tile_pass<1>(h1, nrow0, irow, npsrow, Wt, w_state, lm, half);
    }

    // combine halves (lanes lm and lm+16 hold the same node n)
    partial += __shfl_xor(partial, 16, 32);
    if (lane < 16) sc[wave][lane] = partial;
    __syncthreads();
    if (tid < 16) {
        float tot = 0.f;
        #pragma unroll
        for (int w = 0; w < 8; ++w) tot += sc[w][tid];
        scores[b * N + n0 + tid] = tot;      // deterministic: one writer per (b,n)
    }
}

// ---------------------------------------------------------------------------
// Softmax over N per batch (+ node_mask)
// ---------------------------------------------------------------------------
__global__ __launch_bounds__(512)
void nsm_softmax_kernel(const float* __restrict__ scores,
                        const float* __restrict__ node_mask,
                        float* __restrict__ out)
{
    __shared__ float sh[512];
    const int b = blockIdx.x;
    const int t = threadIdx.x;

    const float v = scores[b * N + t] + node_mask[b * N + t];
    sh[t] = v;
    __syncthreads();
    for (int s = 256; s > 0; s >>= 1) {
        if (t < s) sh[t] = fmaxf(sh[t], sh[t + s]);
        __syncthreads();
    }
    const float mx = sh[0];
    __syncthreads();

    const float e = __expf(v - mx);
    sh[t] = e;
    __syncthreads();
    for (int s = 256; s > 0; s >>= 1) {
        if (t < s) sh[t] += sh[t + s];
        __syncthreads();
    }
    out[b * N + t] = e / sh[0];
}

extern "C" void kernel_launch(void* const* d_in, const int* in_sizes, int n_in,
                              void* d_out, int out_size, void* d_ws, size_t ws_size,
                              hipStream_t stream) {
    const float* node_attr   = (const float*)d_in[0];
    // d_in[1] edge_attr, d_in[3] distribution, d_in[4] ins_id: unused for ins_id = 0
    const float* instruction = (const float*)d_in[2];
    const float* npsim       = (const float*)d_in[5];
    const float* node_mask   = (const float*)d_in[6];
    const float* Wn          = (const float*)d_in[7];
    const float* w_state     = (const float*)d_in[8];

    float*  scores = (float*)d_ws;                         // B*N floats = 64 KB
    __bf16* Wt     = (__bf16*)((char*)d_ws + B * N * 4);   // P*HP*KP bf16 ≈ 1.49 MB

    nsm_prep_kernel<<<(P * HP * KP) / 256, 256, 0, stream>>>(Wn, Wt);
    nsm_gemm_kernel<<<dim3(N / 16, B), 256, 0, stream>>>(
        node_attr, instruction, npsim, Wt, w_state, scores);
    nsm_softmax_kernel<<<dim3(B), 512, 0, stream>>>(scores, node_mask, (float*)d_out);
}
